// SequentialPointNet_5411658793007
// MI455X (gfx1250) — compile-verified, matrix operand order checked by reading, not run on hardware
//
#include <hip/hip_runtime.h>
#include <hip/hip_bf16.h>
#include <stdint.h>

// ---------------------------------------------------------------------------
// Types for CDNA5 WMMA operands
// ---------------------------------------------------------------------------
typedef __bf16 bf16_t;
typedef __attribute__((ext_vector_type(16))) __bf16 v16bf;
typedef __attribute__((ext_vector_type(8)))  __bf16 v8bf;
typedef __attribute__((ext_vector_type(8)))  float  v8f;
typedef __attribute__((ext_vector_type(4)))  float  v4f;
typedef __attribute__((ext_vector_type(2)))  float  v2f;

// ---------------------------------------------------------------------------
// Problem constants (from reference)
// ---------------------------------------------------------------------------
static constexpr int    BF     = 192;                 // B*F frames
static constexpr size_t NPOS1  = (size_t)192*128*64;  // stage-1 positions = 1,572,864
static constexpr size_t NPOS2  = (size_t)192*64*32;   // stage-2 positions =   393,216

// Workspace arena (bytes). Reuse across stages.
static constexpr size_t MB        = 1ull << 20;
static constexpr size_t OFF_FEAT1 = 0;         // f32 [NPOS1][4]              (25.2 MB)
static constexpr size_t OFF_PTS2  = 0;         // f32 [192*128][131]  (reuses feat1)
static constexpr size_t OFF_Y1    = 32*MB;     // bf16 [NPOS1][64]  raw conv out
static constexpr size_t OFF_Y2    = 240*MB;    // bf16 [NPOS1][64]
static constexpr size_t OFF_Y3    = 448*MB;    // bf16 [NPOS1][128]
static constexpr size_t OFF_FEAT2 = 32*MB;     // bf16 [NPOS2][160]  (reuses y1)
static constexpr size_t OFF_Z1    = 240*MB;    // bf16 [NPOS2][128]  (reuses y2)
static constexpr size_t OFF_Z2    = 342*MB;    // bf16 [NPOS2][128]
static constexpr size_t OFF_Z3    = 448*MB;    // bf16 [NPOS2][256]  (reuses y3)
static constexpr size_t OFF_STATS = 851*MB;    // 6 slots * 1024 f32 (sum,sq,scale,shift)
static constexpr size_t OFF_WFRAG = 852*MB;    // 5 slots * 64K bf16 weight fragments
static constexpr size_t WF_SLOT   = 65536;     // bf16 elements per fragment slot

// ---------------------------------------------------------------------------
// Wave helpers (wave32)
// ---------------------------------------------------------------------------
__device__ __forceinline__ unsigned long long shflxor64(unsigned long long v, int m) {
  unsigned lo = (unsigned)v, hi = (unsigned)(v >> 32);
  lo = __shfl_xor(lo, m, 32);
  hi = __shfl_xor(hi, m, 32);
  return ((unsigned long long)hi << 32) | lo;
}
__device__ __forceinline__ unsigned long long wavemin64(unsigned long long v) {
  #pragma unroll
  for (int m = 16; m > 0; m >>= 1) {
    unsigned long long o = shflxor64(v, m);
    v = (o < v) ? o : v;
  }
  return v;
}

// CDNA5 async global->LDS copy (16B) + wait, per CDNA5 ISA
__device__ __forceinline__ void async_copy_b128(unsigned lds_byte_off, const void* gptr) {
  asm volatile("global_load_async_to_lds_b128 %0, %1, off"
               :: "v"(lds_byte_off), "v"(gptr) : "memory");
}
__device__ __forceinline__ void wait_asynccnt0() {
  asm volatile("s_wait_asynccnt 0x0" ::: "memory");
}

// ---------------------------------------------------------------------------
// Misc small kernels
// ---------------------------------------------------------------------------
__global__ void zero_kernel(float* p, int n) {
  int i = blockIdx.x * blockDim.x + threadIdx.x;
  if (i < n) p[i] = 0.f;
}

// ---------------------------------------------------------------------------
// Stage-1 grouping: one wave per (frame, centroid). 512 pts, top-64 by d2.
// feat1[pos][4] = {relx, rely, relz, d2}, f32.
// ---------------------------------------------------------------------------
__global__ __launch_bounds__(128) void group1_kernel(const float* __restrict__ pts,
                                                     float* __restrict__ feat1) {
  const int lane  = threadIdx.x & 31;
  const int wid   = threadIdx.x >> 5;
  const int gid   = blockIdx.x * 4 + wid;       // 0 .. 192*128-1
  const int frame = gid >> 7;
  const int cen   = gid & 127;
  const float* p  = pts + (size_t)frame * 512 * 3;
  const float cx = p[cen*3+0], cy = p[cen*3+1], cz = p[cen*3+2];

  unsigned long long key[16];
  #pragma unroll
  for (int t = 0; t < 16; t++) {
    int idx = lane + 32 * t;
    float dx = p[idx*3+0]-cx, dy = p[idx*3+1]-cy, dz = p[idx*3+2]-cz;
    float d2 = dx*dx + dy*dy + dz*dz;
    key[t] = ((unsigned long long)__float_as_uint(d2) << 32) | (unsigned)idx;
  }
  // per-lane ascending sort (d2, idx) -- matches jax top_k tie-breaking
  for (int i = 1; i < 16; i++) {
    unsigned long long kv = key[i]; int j = i - 1;
    while (j >= 0 && key[j] > kv) { key[j+1] = key[j]; j--; }
    key[j+1] = kv;
  }

  __shared__ int nb[4][64];
  int h = 0;
  for (int r = 0; r < 64; r++) {
    unsigned long long myk = (h < 16) ? key[h] : ~0ull;
    unsigned long long mn  = wavemin64(myk);
    if (myk == mn) h++;                       // unique winner pops
    if (lane == 0) {
      float d2v = __uint_as_float((unsigned)(mn >> 32));
      int   idx = (int)(unsigned)(mn & 0xffffffffu);
      nb[wid][r] = (d2v > 0.06f) ? cen : idx; // radius replacement (R1)
    }
  }
  __syncthreads();

  for (int q = lane; q < 64; q += 32) {
    int j = nb[wid][q];
    float gx = p[j*3+0]-cx, gy = p[j*3+1]-cy, gz = p[j*3+2]-cz;
    float d  = gx*gx + gy*gy + gz*gz;
    size_t pos = (size_t)gid * 64 + q;
    v4f o = {gx, gy, gz, d};
    *(v4f*)(feat1 + pos * 4) = o;
  }
}

// ---------------------------------------------------------------------------
// Stage-2 grouping: one wave per (frame, centroid2). 128 pts (C=131), top-32.
// feat2[pos][160] bf16: [rel(3), feats(128), d2, pad(28)=0]
// ---------------------------------------------------------------------------
__global__ __launch_bounds__(128) void group2_kernel(const float* __restrict__ pts2,
                                                     bf16_t* __restrict__ feat2) {
  const int lane  = threadIdx.x & 31;
  const int wid   = threadIdx.x >> 5;
  const int gid   = blockIdx.x * 4 + wid;       // 0 .. 192*64-1
  const int frame = gid >> 6;
  const int cen   = gid & 63;
  const float* p  = pts2 + (size_t)frame * 128 * 131;
  const float cx = p[cen*131+0], cy = p[cen*131+1], cz = p[cen*131+2];

  unsigned long long key[4];
  #pragma unroll
  for (int t = 0; t < 4; t++) {
    int idx = lane + 32 * t;
    float dx = p[idx*131+0]-cx, dy = p[idx*131+1]-cy, dz = p[idx*131+2]-cz;
    float d2 = dx*dx + dy*dy + dz*dz;
    key[t] = ((unsigned long long)__float_as_uint(d2) << 32) | (unsigned)idx;
  }
  for (int i = 1; i < 4; i++) {
    unsigned long long kv = key[i]; int j = i - 1;
    while (j >= 0 && key[j] > kv) { key[j+1] = key[j]; j--; }
    key[j+1] = kv;
  }

  __shared__ int nb[4][32];
  int h = 0;
  for (int r = 0; r < 32; r++) {
    unsigned long long myk = (h < 4) ? key[h] : ~0ull;
    unsigned long long mn  = wavemin64(myk);
    if (myk == mn) h++;
    if (lane == 0) {
      float d2v = __uint_as_float((unsigned)(mn >> 32));
      int   idx = (int)(unsigned)(mn & 0xffffffffu);
      nb[wid][r] = (d2v > 0.11f) ? cen : idx;  // R2
    }
  }
  __syncthreads();

  for (int q = 0; q < 32; q++) {
    int j = nb[wid][q];
    float gx = p[j*131+0]-cx, gy = p[j*131+1]-cy, gz = p[j*131+2]-cz;
    float d  = gx*gx + gy*gy + gz*gz;
    size_t row = ((size_t)gid * 32 + q) * 160;
    for (int c = lane; c < 160; c += 32) {
      float v;
      if (c < 3)        v = (c == 0) ? gx : ((c == 1) ? gy : gz);
      else if (c < 131) v = p[j*131 + c];
      else if (c == 131) v = d;
      else              v = 0.f;
      feat2[row + c] = (bf16_t)v;
    }
  }
}

// ---------------------------------------------------------------------------
// Layer-1 GEMM, fp32 WMMA 16x16x4 (Cin = 4). One wave per 16-position tile,
// loops over all 4 M-tiles reusing the register-resident B operand.
// ---------------------------------------------------------------------------
__global__ __launch_bounds__(256) void gemm_l1_f32(const float* __restrict__ X4,
                                                   const float* __restrict__ W,
                                                   const float* __restrict__ bias,
                                                   bf16_t* __restrict__ Y,
                                                   float* __restrict__ stats) {
  __shared__ float s_sum[64], s_sq[64];
  if (threadIdx.x < 64) { s_sum[threadIdx.x] = 0.f; s_sq[threadIdx.x] = 0.f; }
  __syncthreads();

  const int lane = threadIdx.x & 31;
  const int wid  = threadIdx.x >> 5;
  const int hi   = lane >> 4;
  const size_t pos = ((size_t)blockIdx.x * 8 + wid) * 16 + (lane & 15);

  // B (4x16 f32): K split per half-wave, column = lane&15 (loaded once)
  v2f b = *(const v2f*)(X4 + pos * 4 + 2 * hi);

  #pragma unroll
  for (int mt = 0; mt < 4; mt++) {
    const int mA = mt * 16 + (lane & 15);
    // A (16x4 f32): lanes 0-15 -> K0,K1 ; lanes 16-31 -> K2,K3
    v2f a = *(const v2f*)(W + (size_t)mA * 4 + 2 * hi);
    v8f c = {};
    c = __builtin_amdgcn_wmma_f32_16x16x4_f32(false, a, false, b, (short)0, c, false, false);

    const int mbase = mt * 16 + hi * 8;
    v8bf yo;
    #pragma unroll
    for (int r = 0; r < 8; r++) {
      float v = c[r] + bias[mbase + r];
      yo[r] = (bf16_t)v;
      atomicAdd(&s_sum[mbase + r], v);
      atomicAdd(&s_sq [mbase + r], v * v);
    }
    *(v8bf*)(Y + pos * 64 + mbase) = yo;
  }

  __syncthreads();
  if (threadIdx.x < 64) {
    atomicAdd(&stats[      threadIdx.x], s_sum[threadIdx.x]);
    atomicAdd(&stats[256 + threadIdx.x], s_sq [threadIdx.x]);
  }
}

// ---------------------------------------------------------------------------
// Generic bf16 GEMM, fully unrolled: KS K-tiles of 32, MT M-tiles of 16.
// - Weights async-staged into LDS once per block (global_load_async_to_lds).
// - B fragments loaded once per wave into registers, reused across all MT
//   (cuts HBM activation traffic by MT vs the naive grid.y scheme).
// - Optional fused BatchNorm-affine + ReLU of the PREVIOUS layer on the
//   B-load path (X holds raw conv output; prev_stats holds scale/shift).
// - Epilogue: bias add, bf16 store of raw conv out, BN sum/sumsq partials
//   via LDS float atomics, one global atomic per channel per block.
// ---------------------------------------------------------------------------
template<int KS, int MT, int COUT, bool BN_PREV>
__global__ __launch_bounds__(256) void gemm_bf16_t(const bf16_t* __restrict__ X,
                                                   const bf16_t* __restrict__ Wfrag,
                                                   const float* __restrict__ bias,
                                                   const float* __restrict__ prev_stats,
                                                   bf16_t* __restrict__ Y,
                                                   float* __restrict__ stats) {
  constexpr int CINPAD = KS * 32;
  __shared__ float  s_sum[COUT], s_sq[COUT];
  __shared__ bf16_t sw[MT * KS * 512];

  // async-stage weight fragments into LDS (16B chunks; NVEC multiple of 256)
  constexpr int NVEC = MT * KS * 512 / 8;
  #pragma unroll 1
  for (int t = threadIdx.x; t < NVEC; t += 256) {
    async_copy_b128((unsigned)(uintptr_t)(&sw[t * 8]), Wfrag + t * 8);
  }
  if (threadIdx.x < COUT) { s_sum[threadIdx.x] = 0.f; s_sq[threadIdx.x] = 0.f; }

  const int lane = threadIdx.x & 31;
  const int wid  = threadIdx.x >> 5;
  const int hi   = lane >> 4;
  const size_t pos = ((size_t)blockIdx.x * 8 + wid) * 16 + (lane & 15);

  // Load all B fragments once (optionally applying prev-layer BN + ReLU)
  v16bf b[KS];
  const bf16_t* xrow = X + pos * (size_t)CINPAD;
  #pragma unroll
  for (int ks = 0; ks < KS; ks++) {
    const int c0 = ks * 32 + hi * 8;
    v8bf b0 = *(const v8bf*)(xrow + c0);
    v8bf b1 = *(const v8bf*)(xrow + c0 + 16);
    if (BN_PREV) {
      v8f sc0 = *(const v8f*)(prev_stats + 512 + c0);
      v8f sh0 = *(const v8f*)(prev_stats + 768 + c0);
      v8f sc1 = *(const v8f*)(prev_stats + 512 + c0 + 16);
      v8f sh1 = *(const v8f*)(prev_stats + 768 + c0 + 16);
      #pragma unroll
      for (int j = 0; j < 8; j++) {
        float x0 = (float)b0[j] * sc0[j] + sh0[j]; x0 = x0 > 0.f ? x0 : 0.f;
        float x1 = (float)b1[j] * sc1[j] + sh1[j]; x1 = x1 > 0.f ? x1 : 0.f;
        b0[j] = (bf16_t)x0; b1[j] = (bf16_t)x1;
      }
    }
    #pragma unroll
    for (int j = 0; j < 8; j++) { b[ks][j] = b0[j]; b[ks][j + 8] = b1[j]; }
  }

  wait_asynccnt0();    // our async copies done
  __syncthreads();     // all waves' copies visible

  // All M-tiles with register-resident B, A from LDS
  #pragma unroll
  for (int mt = 0; mt < MT; mt++) {
    const bf16_t* wb = &sw[(mt * KS * 32 + lane) * 16];
    v8f c = {};
    #pragma unroll
    for (int ks = 0; ks < KS; ks++) {
      v16bf a = *(const v16bf*)(wb + ks * 512);
      c = __builtin_amdgcn_wmma_f32_16x16x32_bf16(false, a, false, b[ks],
                                                  (short)0, c, false, false);
    }
    const int mbase = mt * 16 + hi * 8;
    v8bf yo;
    #pragma unroll
    for (int r = 0; r < 8; r++) {
      float v = c[r] + bias[mbase + r];
      yo[r] = (bf16_t)v;
      atomicAdd(&s_sum[mbase + r], v);
      atomicAdd(&s_sq [mbase + r], v * v);
    }
    *(v8bf*)(Y + pos * (size_t)COUT + mbase) = yo;
  }

  __syncthreads();
  if (threadIdx.x < COUT) {
    atomicAdd(&stats[      threadIdx.x], s_sum[threadIdx.x]);
    atomicAdd(&stats[256 + threadIdx.x], s_sq [threadIdx.x]);
  }
}

// ---------------------------------------------------------------------------
// Weight prep: W (Cout x Cin, f32) -> per-lane A fragments in ISA layout:
//   elem j of lane's v16bf: k = ((j>>3)*2 + (lane>>4))*8 + (j&7)  (+ 32*ks)
//   row m = mtile*16 + (lane&15); zero-pad k >= Cin.
// ---------------------------------------------------------------------------
__global__ void wprep_kernel(const float* __restrict__ W, bf16_t* __restrict__ frag,
                             int Cout, int Cin, int ksteps) {
  int total = (Cout / 16) * ksteps * 512;
  for (int id = blockIdx.x * blockDim.x + threadIdx.x; id < total;
       id += gridDim.x * blockDim.x) {
    int j     = id & 15;
    int lane  = (id >> 4) & 31;
    int blk   = id >> 9;               // mtile*ksteps + ks
    int ks    = blk % ksteps;
    int mtile = blk / ksteps;
    int m = mtile * 16 + (lane & 15);
    int k = ks * 32 + ((j >> 3) * 2 + (lane >> 4)) * 8 + (j & 7);
    float v = (k < Cin) ? W[(size_t)m * Cin + k] : 0.f;
    frag[id] = (bf16_t)v;
  }
}

// ---------------------------------------------------------------------------
// BatchNorm finalize: per-channel scale/shift from global sum/sumsq.
// ---------------------------------------------------------------------------
__global__ void bn_finalize_kernel(float* stats, const float* __restrict__ g,
                                   const float* __restrict__ be, int Cout, float invN) {
  int c = blockIdx.x * blockDim.x + threadIdx.x;
  if (c >= Cout) return;
  float mean = stats[c] * invN;
  float var  = stats[256 + c] * invN - mean * mean;
  var = var > 0.f ? var : 0.f;
  float sc = g[c] * rsqrtf(var + 1e-5f);
  stats[512 + c] = sc;
  stats[768 + c] = be[c] - mean * sc;
}

// ---------------------------------------------------------------------------
// Max over 64 neighbors (BN3+ReLU fused) + build pts2 (xyz + 128 feats).
// ---------------------------------------------------------------------------
__global__ __launch_bounds__(128) void maxpool1_kernel(const bf16_t* __restrict__ y3,
                                                       const float* __restrict__ stats,
                                                       const float* __restrict__ pts,
                                                       float* __restrict__ pts2) {
  int gid   = blockIdx.x;               // frame*128 + cen
  int frame = gid >> 7;
  int cen   = gid & 127;
  int c     = threadIdx.x;              // 0..127
  const float sc = stats[512 + c], sh = stats[768 + c];
  const bf16_t* base = y3 + ((size_t)gid * 64) * 128;
  float m = 0.f;                        // post-ReLU values are >= 0
  for (int q = 0; q < 64; q++) {
    float v = (float)base[(size_t)q * 128 + c] * sc + sh;
    v = v > 0.f ? v : 0.f;
    m = v > m ? v : m;
  }
  float* row = pts2 + (size_t)gid * 131;
  row[3 + c] = m;
  if (c < 3) row[c] = pts[((size_t)frame * 512 + cen) * 3 + c];
}

// ---------------------------------------------------------------------------
// Channel attention: per frame, avg/max over 2048 positions, 132->8->132 MLP,
// sigmoid scale applied to feat2 in place.
// ---------------------------------------------------------------------------
__global__ __launch_bounds__(256) void attention_kernel(bf16_t* feat2,
                                                        const float* __restrict__ fc1,
                                                        const float* __restrict__ fc2) {
  __shared__ float savg[132], smax[132], ha[8], hb[8], sscale[132];
  int frame = blockIdx.x;
  int tid   = threadIdx.x;
  bf16_t* base = feat2 + (size_t)frame * 2048 * 160;

  if (tid < 132) {
    float s = 0.f, m = -3.4e38f;
    for (int p = 0; p < 2048; p++) {
      float v = (float)base[(size_t)p * 160 + tid];
      s += v; m = v > m ? v : m;
    }
    savg[tid] = s * (1.f / 2048.f);
    smax[tid] = m;
  }
  __syncthreads();
  if (tid < 8) {
    float sa = 0.f, sb = 0.f;
    for (int c = 0; c < 132; c++) {
      float w = fc1[tid * 132 + c];
      sa += w * savg[c];
      sb += w * smax[c];
    }
    ha[tid] = sa > 0.f ? sa : 0.f;
    hb[tid] = sb > 0.f ? sb : 0.f;
  }
  __syncthreads();
  if (tid < 132) {
    float o = 0.f;
    for (int k = 0; k < 8; k++) o += fc2[tid * 8 + k] * (ha[k] + hb[k]);
    sscale[tid] = 1.f / (1.f + expf(-o));
  }
  __syncthreads();
  for (int i = tid; i < 2048 * 160; i += blockDim.x) {
    int c = i % 160;
    if (c < 132) {
      float v = (float)base[i] * sscale[c];
      base[i] = (bf16_t)v;
    }
  }
}

// ---------------------------------------------------------------------------
// Final max over 32 neighbors (BN+ReLU of stage-2 layer-3 fused) -> d_out.
// Output layout (Bf, 256, 64, 1) f32.
// ---------------------------------------------------------------------------
__global__ __launch_bounds__(256) void final_max_kernel(const bf16_t* __restrict__ z3,
                                                        const float* __restrict__ stats,
                                                        float* __restrict__ out) {
  int gid   = blockIdx.x;               // frame*64 + cen
  int frame = gid >> 6;
  int cen   = gid & 63;
  int c     = threadIdx.x;              // 0..255
  const float sc = stats[512 + c], sh = stats[768 + c];
  const bf16_t* base = z3 + ((size_t)gid * 32) * 256;
  float m = 0.f;                        // post-ReLU values are >= 0
  for (int q = 0; q < 32; q++) {
    float v = (float)base[(size_t)q * 256 + c] * sc + sh;
    v = v > 0.f ? v : 0.f;
    m = v > m ? v : m;
  }
  out[((size_t)frame * 256 + c) * 64 + cen] = m;
}

// ---------------------------------------------------------------------------
// Host driver
// ---------------------------------------------------------------------------
extern "C" void kernel_launch(void* const* d_in, const int* in_sizes, int n_in,
                              void* d_out, int out_size, void* d_ws, size_t ws_size,
                              hipStream_t stream) {
  (void)in_sizes; (void)n_in; (void)out_size; (void)ws_size;
  const float* clips = (const float*)d_in[0];
  // s1_params: (w,b,g,be) x3
  const float *w1 = (const float*)d_in[1],  *b1 = (const float*)d_in[2];
  const float *g1 = (const float*)d_in[3],  *e1 = (const float*)d_in[4];
  const float *w2 = (const float*)d_in[5],  *b2 = (const float*)d_in[6];
  const float *g2 = (const float*)d_in[7],  *e2 = (const float*)d_in[8];
  const float *w3 = (const float*)d_in[9],  *b3 = (const float*)d_in[10];
  const float *g3 = (const float*)d_in[11], *e3 = (const float*)d_in[12];
  const float *fc1 = (const float*)d_in[13], *fc2 = (const float*)d_in[14];
  // s2_params
  const float *u1 = (const float*)d_in[15], *c1 = (const float*)d_in[16];
  const float *h1 = (const float*)d_in[17], *f1 = (const float*)d_in[18];
  const float *u2 = (const float*)d_in[19], *c2 = (const float*)d_in[20];
  const float *h2 = (const float*)d_in[21], *f2 = (const float*)d_in[22];
  const float *u3 = (const float*)d_in[23], *c3 = (const float*)d_in[24];
  const float *h3 = (const float*)d_in[25], *f3 = (const float*)d_in[26];

  char* ws = (char*)d_ws;
  float*  feat1 = (float*) (ws + OFF_FEAT1);
  float*  pts2  = (float*) (ws + OFF_PTS2);
  bf16_t* y1    = (bf16_t*)(ws + OFF_Y1);
  bf16_t* y2    = (bf16_t*)(ws + OFF_Y2);
  bf16_t* y3    = (bf16_t*)(ws + OFF_Y3);
  bf16_t* feat2 = (bf16_t*)(ws + OFF_FEAT2);
  bf16_t* z1    = (bf16_t*)(ws + OFF_Z1);
  bf16_t* z2    = (bf16_t*)(ws + OFF_Z2);
  bf16_t* z3    = (bf16_t*)(ws + OFF_Z3);
  float*  stats = (float*) (ws + OFF_STATS);   // 6 slots of 1024 f32
  bf16_t* wfrag = (bf16_t*)(ws + OFF_WFRAG);   // 5 slots of WF_SLOT bf16

  const float invN1 = 1.f / (float)NPOS1;
  const float invN2 = 1.f / (float)NPOS2;
  const int NT1 = (int)(NPOS1 / 16 / 8);       // 12288 blocks (8 waves each)
  const int NT2 = (int)(NPOS2 / 16 / 8);       //  3072 blocks

  // 0) zero BN stats
  zero_kernel<<<24, 256, 0, stream>>>(stats, 6 * 1024);

  // 1) stage-1 grouping
  group1_kernel<<<BF * 128 / 4, 128, 0, stream>>>(clips, feat1);

  // 2) stage-1 layer 1: fp32 WMMA (Cin=4), Cout=64, raw output + stats
  gemm_l1_f32<<<NT1, 256, 0, stream>>>(feat1, w1, b1, y1, stats + 0 * 1024);
  bn_finalize_kernel<<<1, 64, 0, stream>>>(stats + 0 * 1024, g1, e1, 64, invN1);

  // 3) stage-1 layer 2: 64->64 (BN1+ReLU fused on load)
  wprep_kernel<<<16, 256, 0, stream>>>(w2, wfrag + 0 * WF_SLOT, 64, 64, 2);
  gemm_bf16_t<2, 4, 64, true><<<NT1, 256, 0, stream>>>(
      y1, wfrag + 0 * WF_SLOT, b2, stats + 0 * 1024, y2, stats + 1 * 1024);
  bn_finalize_kernel<<<1, 64, 0, stream>>>(stats + 1 * 1024, g2, e2, 64, invN1);

  // 4) stage-1 layer 3: 64->128 (BN2+ReLU fused on load)
  wprep_kernel<<<32, 256, 0, stream>>>(w3, wfrag + 1 * WF_SLOT, 128, 64, 2);
  gemm_bf16_t<2, 8, 128, true><<<NT1, 256, 0, stream>>>(
      y2, wfrag + 1 * WF_SLOT, b3, stats + 1 * 1024, y3, stats + 2 * 1024);
  bn_finalize_kernel<<<1, 128, 0, stream>>>(stats + 2 * 1024, g3, e3, 128, invN1);

  // 5) neighbor max-pool (BN3+ReLU fused) + build 131-channel point cloud
  maxpool1_kernel<<<BF * 128, 128, 0, stream>>>(y3, stats + 2 * 1024, clips, pts2);

  // 6) stage-2 grouping (C=131 -> 132 feat channels, padded to 160)
  group2_kernel<<<BF * 64 / 4, 128, 0, stream>>>(pts2, feat2);

  // 7) channel attention (in place on feat2)
  attention_kernel<<<BF, 256, 0, stream>>>(feat2, fc1, fc2);

  // 8) stage-2 layer 1: 132(pad160)->128, no BN on input (feat2 is final)
  wprep_kernel<<<80, 256, 0, stream>>>(u1, wfrag + 2 * WF_SLOT, 128, 132, 5);
  gemm_bf16_t<5, 8, 128, false><<<NT2, 256, 0, stream>>>(
      feat2, wfrag + 2 * WF_SLOT, c1, stats + 3 * 1024, z1, stats + 3 * 1024);
  bn_finalize_kernel<<<1, 128, 0, stream>>>(stats + 3 * 1024, h1, f1, 128, invN2);

  // 9) stage-2 layer 2: 128->128 (BN fused)
  wprep_kernel<<<64, 256, 0, stream>>>(u2, wfrag + 3 * WF_SLOT, 128, 128, 4);
  gemm_bf16_t<4, 8, 128, true><<<NT2, 256, 0, stream>>>(
      z1, wfrag + 3 * WF_SLOT, c2, stats + 3 * 1024, z2, stats + 4 * 1024);
  bn_finalize_kernel<<<1, 128, 0, stream>>>(stats + 4 * 1024, h2, f2, 128, invN2);

  // 10) stage-2 layer 3: 128->256 (BN fused)
  wprep_kernel<<<128, 256, 0, stream>>>(u3, wfrag + 4 * WF_SLOT, 256, 128, 4);
  gemm_bf16_t<4, 16, 256, true><<<NT2, 256, 0, stream>>>(
      z2, wfrag + 4 * WF_SLOT, c3, stats + 4 * 1024, z3, stats + 5 * 1024);
  bn_finalize_kernel<<<1, 256, 0, stream>>>(stats + 5 * 1024, h3, f3, 256, invN2);

  // 11) final neighbor max (BN+ReLU fused) -> d_out (Bf,256,64,1) f32
  final_max_kernel<<<BF * 64, 256, 0, stream>>>(z3, stats + 5 * 1024, (float*)d_out);
}
